// EdgeDecoder_25357486915897
// MI455X (gfx1250) — matrix operand order repaired and checked by reading, not compiled, not run on hardware
//
#include <hip/hip_runtime.h>

typedef __attribute__((ext_vector_type(2)))  float     v2f;
typedef __attribute__((ext_vector_type(8)))  float     v8f;
typedef __attribute__((ext_vector_type(16))) _Float16  v16h;
typedef __attribute__((ext_vector_type(4)))  int       v4i;

#define BB    4
#define NN    256
#define DD    256
#define EE    64
#define KPF   260   // padded f32 LDS row stride (1040 B, 16B-aligned, 4-bank shift)
#define KPH   264   // padded f16 LDS row stride (528 B, 16B-aligned, 4-bank shift)
#define ZROWS 128   // 2 i * 64 j rows of z per block

#if __has_builtin(__builtin_amdgcn_global_load_async_to_lds_b128)
#define HAVE_ASYNC_LDS 1
using gas_v4i = __attribute__((address_space(1))) v4i;   // global
using las_v4i = __attribute__((address_space(3))) v4i;   // LDS
#endif

// 16-byte global -> LDS copy: async direct-to-LDS on gfx1250 if available.
__device__ __forceinline__ void async_copy16(const void* g, void* l) {
#ifdef HAVE_ASYNC_LDS
    __builtin_amdgcn_global_load_async_to_lds_b128(
        (gas_v4i*)g, (las_v4i*)l, 0, 0);
#else
    *(float4*)l = *(const float4*)g;
#endif
}

__device__ __forceinline__ void async_wait0() {
#ifdef HAVE_ASYNC_LDS
    asm volatile("s_wait_asynccnt 0" ::: "memory");
#endif
}

// ---------------------------------------------------------------------------
// Generic f32 WMMA GEMM:  C[M x 256] = A[M x 256] @ W[n, ldw](cols koff..)^T (+bias)
// Block = 256 threads (8 waves), tile 64(M) x 32(N), K = 256 in steps of 4.
// ---------------------------------------------------------------------------
__global__ __launch_bounds__(256) void gemm_wmma_f32(
    const float* __restrict__ A, const float* __restrict__ W, int ldw, int koff,
    const float* __restrict__ bias, float* __restrict__ C)
{
    __shared__ float As[64 * KPF];
    __shared__ float Ws[32 * KPF];

    const int m0  = blockIdx.x * 64;
    const int n0  = blockIdx.y * 32;
    const int tid = threadIdx.x;

    // Stage tiles into LDS with async 16B/lane copies (4096 + 2048 chunks).
    for (int idx = tid; idx < 64 * 64; idx += 256) {
        int r = idx >> 6, c = (idx & 63) << 2;
        async_copy16(&A[(m0 + r) * 256 + c], &As[r * KPF + c]);
    }
    for (int idx = tid; idx < 32 * 64; idx += 256) {
        int r = idx >> 6, c = (idx & 63) << 2;
        async_copy16(&W[(n0 + r) * ldw + koff + c], &Ws[r * KPF + c]);
    }
    async_wait0();
    __syncthreads();

    const int lane = tid & 31;
    const int wv   = tid >> 5;
    const int mt   = (wv & 3) * 16;
    const int nt   = (wv >> 2) * 16;
    const int rsel = lane & 15;
    const int ksel = (lane >> 4) * 2;

    v8f acc = {0.f, 0.f, 0.f, 0.f, 0.f, 0.f, 0.f, 0.f};
    for (int k0 = 0; k0 < 256; k0 += 4) {
        const int kc = k0 + ksel;
        v2f a, b;
        a.x = As[(mt + rsel) * KPF + kc];
        a.y = As[(mt + rsel) * KPF + kc + 1];
        b.x = Ws[(nt + rsel) * KPF + kc];
        b.y = Ws[(nt + rsel) * KPF + kc + 1];
        acc = __builtin_amdgcn_wmma_f32_16x16x4_f32(
                  false, a, false, b, (short)0, acc, false, false);
    }

    const int   n  = n0 + nt + rsel;
    const float bv = bias ? bias[n] : 0.f;
    const int   mr = m0 + mt + (lane >> 4) * 8;
#pragma unroll
    for (int r = 0; r < 8; ++r)
        C[(mr + r) * 256 + n] = acc[r] + bv;
}

// ---------------------------------------------------------------------------
// One-time W2 f32 -> f16 conversion (64x256 = 16384 elements)
// ---------------------------------------------------------------------------
__global__ __launch_bounds__(256) void w2_to_half(
    const float* __restrict__ W2, _Float16* __restrict__ W2h)
{
    const int idx = blockIdx.x * 256 + threadIdx.x;
    W2h[idx] = (_Float16)W2[idx];
}

// ---------------------------------------------------------------------------
// Fused edge kernel: out[b,i,j,e] = silu(pa[b,i,:]+pb[b,j,:]+b1) . W2[e,:] + b2[e]
// Block = 256 threads, owns (b, 2 i's, 64 j's, all 64 e).
// Wave = one 16-row M-subtile, 4 accumulators (e-tiles), A-frag reused 4x.
// ---------------------------------------------------------------------------
__global__ __launch_bounds__(256) void edge_out_wmma(
    const float* __restrict__ pa, const float* __restrict__ pb,
    const float* __restrict__ b1, const _Float16* __restrict__ W2h,
    const float* __restrict__ b2, float* __restrict__ out)
{
    __shared__ _Float16 zsh[ZROWS * KPH];   // silu(z) tile, f16 (128 rows)
    __shared__ _Float16 w2h[64 * KPH];      // W2, f16
    __shared__ float    pab[2 * 256];       // two pa rows + b1

    const int bid  = blockIdx.x;
    const int jblk = bid & 3;               // N/64 = 4
    const int iblk = (bid >> 2) & 127;      // N/2  = 128
    const int b    = bid >> 9;
    const int i0   = iblk * 2;
    const int j0   = jblk * 64;
    const int tid  = threadIdx.x;

    // W2h -> LDS via async 16B copies (64 rows x 32 chunks of 8 halves)
    for (int idx = tid; idx < 64 * 32; idx += 256) {
        int e = idx >> 5, c = (idx & 31) << 3;
        async_copy16(&W2h[e * 256 + c], &w2h[e * KPH + c]);
    }
    // pa rows + b1 (needs VALU add -> regular path)
    for (int idx = tid; idx < 2 * 256; idx += 256) {
        int il = idx >> 8, k = idx & 255;
        pab[idx] = pa[((b * NN) + i0 + il) * DD + k] + b1[k];
    }
    async_wait0();
    __syncthreads();

    // z tile: 128 rows x 256, SiLU computed once per element
    for (int idx = tid; idx < ZROWS * 256; idx += 256) {
        int row = idx >> 8, k = idx & 255;
        int il  = row >> 6, jl = row & 63;
        float v = pab[(il << 8) + k] + pb[((b * NN) + j0 + jl) * DD + k];
        float s = __builtin_amdgcn_rcpf(1.0f + __expf(-v));
        zsh[row * KPH + k] = (_Float16)(v * s);
    }
    __syncthreads();

    const int lane = tid & 31;
    const int wv   = tid >> 5;              // M-subtile: il = wv>>2, jsub = wv&3
    const int rsel = lane & 15;
    const int hi   = lane >> 4;
    const int arow = (wv << 4) + rsel;      // row in zsh (il*64 + jsub*16 + rsel)

    const v8f vzero = {0.f, 0.f, 0.f, 0.f, 0.f, 0.f, 0.f, 0.f};
    v8f acc[4];
#pragma unroll
    for (int t = 0; t < 4; ++t) acc[t] = vzero;

    for (int kb = 0; kb < 256; kb += 32) {
        // A fragment (16-bit A 16x32): two contiguous 8-half runs per lane
        v16h a;
        const _Float16* za = &zsh[arow * KPH + kb + 8 * hi];
#pragma unroll
        for (int t = 0; t < 8; ++t) a[t] = za[t];
#pragma unroll
        for (int t = 0; t < 8; ++t) a[8 + t] = za[16 + t];

#pragma unroll
        for (int et = 0; et < 4; ++et) {
            // B fragment (16-bit B 32x16): two contiguous 8-half runs per lane
            v16h w;
            const _Float16* wa = &w2h[(et * 16 + rsel) * KPH + kb + 8 * hi];
#pragma unroll
            for (int t = 0; t < 8; ++t) w[t] = wa[t];
#pragma unroll
            for (int t = 0; t < 8; ++t) w[8 + t] = wa[16 + t];
            acc[et] = __builtin_amdgcn_wmma_f32_16x16x32_f16(
                          false, a, false, w, (short)0, acc[et], false, false);
        }
    }

    const int i  = i0 + (wv >> 2);
    const int jr = j0 + (wv & 3) * 16 + hi * 8;
#pragma unroll
    for (int et = 0; et < 4; ++et) {
        const int   e   = et * 16 + rsel;
        const float b2v = b2[e];
#pragma unroll
        for (int r = 0; r < 8; ++r)
            out[(((b * NN + i) * NN) + jr + r) * EE + e] = acc[et][r] + b2v;
    }
}

// ---------------------------------------------------------------------------
extern "C" void kernel_launch(void* const* d_in, const int* in_sizes, int n_in,
                              void* d_out, int out_size, void* d_ws, size_t ws_size,
                              hipStream_t stream)
{
    const float* x    = (const float*)d_in[0];
    const float* W_in = (const float*)d_in[1];
    const float* b_in = (const float*)d_in[2];
    const float* W1   = (const float*)d_in[3];
    const float* b1   = (const float*)d_in[4];
    const float* W2   = (const float*)d_in[5];
    const float* b2   = (const float*)d_in[6];
    float*       out  = (float*)d_out;

    float*    ws  = (float*)d_ws;
    float*    h   = ws;                         // 1 MB
    float*    pa  = ws + BB * NN * DD;          // 1 MB
    float*    pb  = ws + 2 * BB * NN * DD;      // 1 MB
    _Float16* W2h = (_Float16*)(ws + 3 * BB * NN * DD);  // 32 KB

    dim3 g1(BB * NN / 64, DD / 32);             // (16, 8)
    gemm_wmma_f32<<<g1, 256, 0, stream>>>(x, W_in, 256,   0, b_in,   h);
    gemm_wmma_f32<<<g1, 256, 0, stream>>>(h, W1,   512,   0, nullptr, pa);
    gemm_wmma_f32<<<g1, 256, 0, stream>>>(h, W1,   512, 256, nullptr, pb);
    w2_to_half  <<<EE * DD / 256, 256, 0, stream>>>(W2, W2h);

    edge_out_wmma<<<BB * (NN / 2) * (NN / 64), 256, 0, stream>>>(
        pa, pb, b1, W2h, b2, out);
}
